// FeatureExtractor_1_5136780886319
// MI455X (gfx1250) — matrix operand order repaired and checked by reading, not compiled, not run on hardware
//
#include <hip/hip_runtime.h>
#include <hip/hip_bf16.h>

// ---------------------------------------------------------------------------
// MI455X (gfx1250) implementation of the PCT/DGCNN feature extractor.
// All GEMM-shaped work (kNN Gram, edge-conv, q/v/trans projections, q@qT,
// v@att) runs on v_wmma_f32_16x16x32_f16 (f16 in, f32 accumulate).
// Big (Nd=2048) GEMMs use a 32x64 wave tile: 8 WMMAs per K-step from
// 2 A-frags + 4 B-frags (2x arithmetic intensity vs 16x32).
// N x N attention scratch is processed per-batch so it lives in the 192MB L2.
// wave32 everywhere; 128-thread blocks = 4 waves.
// ---------------------------------------------------------------------------

#define NPTS 2048
#define NBATCH 8
#define KNN 20

typedef _Float16 hvec8  __attribute__((ext_vector_type(8)));
typedef _Float16 hvec16 __attribute__((ext_vector_type(16)));
typedef float    fvec8  __attribute__((ext_vector_type(8)));

__device__ __forceinline__ fvec8 wmma_f16(hvec16 a, hvec16 b, fvec8 c) {
  return __builtin_amdgcn_wmma_f32_16x16x32_f16(false, a, false, b, (short)0, c,
                                                false, false);
}

__device__ __forceinline__ hvec16 pack16(hvec8 lo, hvec8 hi) {
  return __builtin_shufflevector(lo, hi, 0, 1, 2, 3, 4, 5, 6, 7, 8, 9, 10, 11,
                                 12, 13, 14, 15);
}

// A fragment (16x32 f16, M = row): lane<16 holds K {0..7,16..23} of row
// (lane&15); lane>=16 holds K {8..15,24..31}.  Two contiguous 16B loads.
__device__ __forceinline__ hvec16 fragA(const _Float16* __restrict__ A, int lda,
                                        int m0, int k0, int lane) {
  const _Float16* p =
      A + (size_t)(m0 + (lane & 15)) * lda + (k0 + ((lane & 16) ? 8 : 0));
  hvec8 lo = *(const hvec8*)p;
  hvec8 hi = *(const hvec8*)(p + 16);
  return pack16(lo, hi);
}

// B fragment (32x16 f16) loaded from BT stored as (Ncols x K) row-major:
// lane<16: column (lane&15), K {0..7,16..23}; lane>=16: K {8..15,24..31}.
__device__ __forceinline__ hvec16 fragB(const _Float16* __restrict__ BT,
                                        int ldb, int n0, int k0, int lane) {
  const _Float16* p =
      BT + (size_t)(n0 + (lane & 15)) * ldb + (k0 + ((lane & 16) ? 8 : 0));
  hvec8 lo = *(const hvec8*)p;
  hvec8 hi = *(const hvec8*)(p + 16);
  return pack16(lo, hi);
}

// ---------------------------------------------------------------------------
// Narrow WMMA GEMM (q projection): D(M x Nd) = A(M x K) * BT(Nd x K)^T.
// Block = 128 threads (4 waves); each wave owns a 16(M) x 32(Nd) strip.
// grid = (M/64, Nd/32, batches).  MODE 0: store f16.
// ---------------------------------------------------------------------------
template <int MODE>
__global__ __launch_bounds__(128) void gemm16(
    const _Float16* __restrict__ A, const _Float16* __restrict__ BT,
    float* __restrict__ outF, _Float16* __restrict__ outH,
    const float* __restrict__ bias, const float* __restrict__ gamma,
    const float* __restrict__ beta, const float* __restrict__ xres, int Kdim,
    int lda, int ldb, int ldd, long long sA, long long sB, long long sO) {
  const int z = blockIdx.z;
  A += (size_t)z * (size_t)sA;
  BT += (size_t)z * (size_t)sB;
  const size_t ob = (size_t)z * (size_t)sO;
  const int lane = threadIdx.x & 31;
  const int m0 = blockIdx.x * 64 + (threadIdx.x >> 5) * 16;
  const int n0 = blockIdx.y * 32;
  fvec8 acc0 = {0.f, 0.f, 0.f, 0.f, 0.f, 0.f, 0.f, 0.f};
  fvec8 acc1 = acc0;
  for (int k0 = 0; k0 < Kdim; k0 += 32) {
    hvec16 a = fragA(A, lda, m0, k0, lane);
    hvec16 b0 = fragB(BT, ldb, n0, k0, lane);
    hvec16 b1 = fragB(BT, ldb, n0 + 16, k0, lane);
    acc0 = wmma_f16(a, b0, acc0);
    acc1 = wmma_f16(a, b1, acc1);
  }
  const int rb = m0 + ((lane >> 4) << 3);  // D layout: row = rb + r
  const int col = n0 + (lane & 15);
#pragma unroll
  for (int r = 0; r < 8; ++r) {
    const int m = rb + r;
    if (MODE == 0) {
      outH[ob + (size_t)m * ldd + col] = (_Float16)acc0[r];
      outH[ob + (size_t)m * ldd + col + 16] = (_Float16)acc1[r];
    } else {
      outF[ob + (size_t)m * ldd + col] = acc0[r];
      outF[ob + (size_t)m * ldd + col + 16] = acc1[r];
    }
  }
  (void)bias; (void)gamma; (void)beta; (void)xres;
}

// ---------------------------------------------------------------------------
// Wide WMMA GEMM for Nd % 128 == 0:  32(M) x 64(Nd) per wave, 8 WMMAs per
// K-step from 2 A-frags + 4 B-frags.  Block tile = 64(M) x 128(Nd), 4 waves.
// grid = (M/64, Nd/128, batches).
// MODE 1: store f32            MODE 2: +bias[row], store f16
// MODE 3: residual epilogue out[n][o] = x[n][o] + relu((d+tb)*g+b), f32+f16
// ---------------------------------------------------------------------------
template <int MODE>
__global__ __launch_bounds__(128) void gemm_big(
    const _Float16* __restrict__ A, const _Float16* __restrict__ BT,
    float* __restrict__ outF, _Float16* __restrict__ outH,
    const float* __restrict__ bias, const float* __restrict__ gamma,
    const float* __restrict__ beta, const float* __restrict__ xres, int Kdim,
    int lda, int ldb, int ldd, long long sA, long long sB, long long sO) {
  const int z = blockIdx.z;
  A += (size_t)z * (size_t)sA;
  BT += (size_t)z * (size_t)sB;
  const size_t ob = (size_t)z * (size_t)sO;
  const int lane = threadIdx.x & 31;
  const int w = threadIdx.x >> 5;
  const int m0 = blockIdx.x * 64 + (w & 1) * 32;
  const int n0 = blockIdx.y * 128 + (w >> 1) * 64;
  const fvec8 z8 = {0.f, 0.f, 0.f, 0.f, 0.f, 0.f, 0.f, 0.f};
  fvec8 acc[2][4];
#pragma unroll
  for (int i = 0; i < 2; ++i)
#pragma unroll
    for (int j = 0; j < 4; ++j) acc[i][j] = z8;
  for (int k0 = 0; k0 < Kdim; k0 += 32) {
    hvec16 a0 = fragA(A, lda, m0, k0, lane);
    hvec16 a1 = fragA(A, lda, m0 + 16, k0, lane);
#pragma unroll
    for (int j = 0; j < 4; ++j) {
      hvec16 bf = fragB(BT, ldb, n0 + j * 16, k0, lane);
      acc[0][j] = wmma_f16(a0, bf, acc[0][j]);
      acc[1][j] = wmma_f16(a1, bf, acc[1][j]);
    }
  }
#pragma unroll
  for (int i = 0; i < 2; ++i) {
    const int rb = m0 + i * 16 + ((lane >> 4) << 3);
#pragma unroll
    for (int j = 0; j < 4; ++j) {
      const int col = n0 + j * 16 + (lane & 15);
#pragma unroll
      for (int r = 0; r < 8; ++r) {
        const int m = rb + r;
        const float d = acc[i][j][r];
        if (MODE == 1) {
          outF[ob + (size_t)m * ldd + col] = d;
        } else if (MODE == 2) {
          outH[ob + (size_t)m * ldd + col] = (_Float16)(d + bias[m]);
        } else {  // MODE 3
          float t = (d + bias[m]) * gamma[m] + beta[m];
          t = t > 0.f ? t : 0.f;
          const size_t i0 = ob + (size_t)col * ldd + m;
          const float v = xres[i0] + t;
          outF[i0] = v;
          outH[i0] = (_Float16)v;
        }
      }
    }
  }
}

// ---------------------------------------------------------------------------
// WMMA edge conv (layers 2..4): for each point, gather k=20 neighbor edge
// features e=[nbr-ctr, ctr] (2C wide), GEMM against W(O x 2C), then
// leakyrelu(d*g+b) and max over k.  Weight 32-col tile cached in LDS.
// grid = (NPTS/64, O/32, B).
// ---------------------------------------------------------------------------
__global__ __launch_bounds__(128) void edge_conv_wmma(
    const _Float16* __restrict__ xh, const int* __restrict__ idx,
    const _Float16* __restrict__ W, const float* __restrict__ g,
    const float* __restrict__ bb, float* __restrict__ hF,
    _Float16* __restrict__ hH, int C, int O) {
  extern __shared__ _Float16 ldsW[];  // 32 rows x 2C halves
  const int b = blockIdx.z;
  const int C2 = C * 2;
  xh += (size_t)b * NPTS * C;
  idx += (size_t)b * NPTS * KNN;
  hF += (size_t)b * NPTS * O;
  hH += (size_t)b * NPTS * O;
  const int tid = threadIdx.x;
  const int lane = tid & 31;
  const int n0 = blockIdx.y * 32;
  for (int i = tid; i < 32 * C2; i += 128)
    ldsW[i] = W[(size_t)(n0 + i / C2) * C2 + (i % C2)];
  __syncthreads();
  const int m0 = blockIdx.x * 64 + (tid >> 5) * 16;
  const int p = m0 + (lane & 15);
  const _Float16* crow = xh + (size_t)p * C;
  const int hs = (lane & 16) ? 8 : 0;
  const int bs = (lane & 16) ? 8 : 0;
  const int colA = n0 + (lane & 15);
  const float g0 = g[colA], b0f = bb[colA];
  const float g1 = g[colA + 16], b1f = bb[colA + 16];
  fvec8 hm0, hm1;
#pragma unroll
  for (int r = 0; r < 8; ++r) {
    hm0[r] = -3.4e38f;
    hm1[r] = -3.4e38f;
  }
  for (int k = 0; k < KNN; ++k) {
    const int nb = idx[p * KNN + k];
    if (k + 1 < KNN)  // hide next gather behind this k's WMMAs
      __builtin_prefetch(xh + (size_t)idx[p * KNN + k + 1] * C, 0, 1);
    const _Float16* nrow = xh + (size_t)nb * C;
    fvec8 acc0 = {0.f, 0.f, 0.f, 0.f, 0.f, 0.f, 0.f, 0.f};
    fvec8 acc1 = acc0;
    for (int kt = 0; kt < C2; kt += 32) {
      const int s0 = kt + hs, s1 = kt + hs + 16;
      hvec8 e0, e1;
      if (s0 < C)
        e0 = *(const hvec8*)(nrow + s0) - *(const hvec8*)(crow + s0);
      else
        e0 = *(const hvec8*)(crow + (s0 - C));
      if (s1 < C)
        e1 = *(const hvec8*)(nrow + s1) - *(const hvec8*)(crow + s1);
      else
        e1 = *(const hvec8*)(crow + (s1 - C));
      hvec16 a = pack16(e0, e1);
      const _Float16* pb0 = ldsW + (size_t)(lane & 15) * C2 + kt + bs;
      const _Float16* pb1 = ldsW + (size_t)((lane & 15) + 16) * C2 + kt + bs;
      hvec16 f0 = pack16(*(const hvec8*)pb0, *(const hvec8*)(pb0 + 16));
      hvec16 f1 = pack16(*(const hvec8*)pb1, *(const hvec8*)(pb1 + 16));
      acc0 = wmma_f16(a, f0, acc0);
      acc1 = wmma_f16(a, f1, acc1);
    }
#pragma unroll
    for (int r = 0; r < 8; ++r) {
      float v0 = acc0[r] * g0 + b0f;
      v0 = v0 > 0.f ? v0 : 0.2f * v0;
      float v1 = acc1[r] * g1 + b1f;
      v1 = v1 > 0.f ? v1 : 0.2f * v1;
      hm0[r] = fmaxf(hm0[r], v0);
      hm1[r] = fmaxf(hm1[r], v1);
    }
  }
  const int rb = m0 + ((lane >> 4) << 3);
#pragma unroll
  for (int r = 0; r < 8; ++r) {
    const size_t i0 = (size_t)(rb + r) * O + colA;
    hF[i0] = hm0[r];
    hH[i0] = (_Float16)hm0[r];
    hF[i0 + 16] = hm1[r];
    hH[i0 + 16] = (_Float16)hm1[r];
  }
}

// ---------------------------------------------------------------------------
// Layer-1 edge conv (C=3, O=64) - K too small for WMMA, pure VALU.
// ---------------------------------------------------------------------------
__global__ __launch_bounds__(256) void edge_conv1(
    const float* __restrict__ x, const int* __restrict__ idx,
    const float* __restrict__ w, const float* __restrict__ g,
    const float* __restrict__ bb, float* __restrict__ hF,
    _Float16* __restrict__ hH) {
  const int gi = blockIdx.x * 256 + threadIdx.x;
  if (gi >= NBATCH * NPTS * 64) return;
  const int o = gi & 63;
  const int pn = gi >> 6;  // global point id b*2048+n
  const float* ctr = x + (size_t)pn * 3;
  const float c0 = ctr[0], c1 = ctr[1], c2 = ctr[2];
  const float w0 = w[o * 6 + 0], w1 = w[o * 6 + 1], w2 = w[o * 6 + 2];
  const float w3 = w[o * 6 + 3], w4 = w[o * 6 + 4], w5 = w[o * 6 + 5];
  const float cw = c0 * w3 + c1 * w4 + c2 * w5;
  const float cg = g[o], cb = bb[o];
  const int bbase = pn & ~(NPTS - 1);
  float hm = -3.4e38f;
  for (int k = 0; k < KNN; ++k) {
    const int j = idx[pn * KNN + k];
    const float* nb = x + (size_t)(bbase + j) * 3;
    float d = (nb[0] - c0) * w0 + (nb[1] - c1) * w1 + (nb[2] - c2) * w2 + cw;
    d = d * cg + cb;
    d = d > 0.f ? d : 0.2f * d;
    hm = fmaxf(hm, d);
  }
  hF[gi] = hm;
  hH[gi] = (_Float16)hm;
}

// ---------------------------------------------------------------------------
// kNN top-20 from a Gram row: neg_dist = 2*G[i][j] - G[i][i] - G[j][j].
// One wave32 per row: per-lane sorted-20 list, then 20 shuffle-argmax pops.
// ---------------------------------------------------------------------------
__global__ __launch_bounds__(256) void knn_topk(const float* __restrict__ gram,
                                                const float* __restrict__ dg,
                                                int* __restrict__ out) {
  const int lane = threadIdx.x & 31;
  const int i = blockIdx.x * 8 + (threadIdx.x >> 5);
  const float* row = gram + (size_t)i * NPTS;
  const float di = dg[i];
  float val[KNN];
  int vid[KNN];
#pragma unroll
  for (int t = 0; t < KNN; ++t) {
    val[t] = -3.4e38f;
    vid[t] = 0;
  }
  for (int j = lane; j < NPTS; j += 32) {
    const float nd = 2.f * row[j] - di - dg[j];
    if (nd > val[KNN - 1]) {
      val[KNN - 1] = nd;
      vid[KNN - 1] = j;
#pragma unroll
      for (int p = KNN - 1; p > 0; --p) {
        if (val[p] > val[p - 1]) {
          float tv = val[p];
          val[p] = val[p - 1];
          val[p - 1] = tv;
          int ti = vid[p];
          vid[p] = vid[p - 1];
          vid[p - 1] = ti;
        }
      }
    }
  }
  float cur = val[0];
  int curi = vid[0];
  for (int kk = 0; kk < KNN; ++kk) {
    float m = cur;
#pragma unroll
    for (int off = 16; off > 0; off >>= 1) m = fmaxf(m, __shfl_xor(m, off, 32));
    const unsigned long long msk = __ballot(cur == m);
    const int wl = __ffsll(msk) - 1;
    const int wi = __shfl(curi, wl, 32);
    if (lane == 0) out[(size_t)i * KNN + kk] = wi;
    if (lane == wl) {
#pragma unroll
      for (int p = 0; p < KNN - 1; ++p) {
        val[p] = val[p + 1];
        vid[p] = vid[p + 1];
      }
      val[KNN - 1] = -3.4e38f;
      cur = val[0];
      curi = vid[0];
    }
  }
}

__global__ void gram_diag(const float* __restrict__ gram,
                          float* __restrict__ dg) {
  const int j = blockIdx.x * 256 + threadIdx.x;
  if (j < NPTS) dg[j] = gram[(size_t)j * (NPTS + 1)];
}

// row softmax in place (f32, one block per row)
__global__ __launch_bounds__(256) void softmax_row(float* __restrict__ att) {
  __shared__ float red[256];
  float* row = att + (size_t)blockIdx.x * NPTS;
  const int tid = threadIdx.x;
  float m = -3.4e38f;
  for (int j = tid; j < NPTS; j += 256) m = fmaxf(m, row[j]);
  red[tid] = m;
  __syncthreads();
  for (int s = 128; s > 0; s >>= 1) {
    if (tid < s) red[tid] = fmaxf(red[tid], red[tid + s]);
    __syncthreads();
  }
  m = red[0];
  __syncthreads();
  float sum = 0.f;
  for (int j = tid; j < NPTS; j += 256) {
    const float e = __expf(row[j] - m);
    row[j] = e;
    sum += e;
  }
  red[tid] = sum;
  __syncthreads();
  for (int s = 128; s > 0; s >>= 1) {
    if (tid < s) red[tid] += red[tid + s];
    __syncthreads();
  }
  const float inv = 1.f / red[0];
  for (int j = tid; j < NPTS; j += 256) row[j] *= inv;
}

__global__ void col_sum(const float* __restrict__ att, float* __restrict__ cs) {
  const int m = blockIdx.x * 256 + threadIdx.x;
  float s = 0.f;
  for (int n = 0; n < NPTS; ++n) s += att[(size_t)n * NPTS + m];
  cs[m] = s;
}

// attT[m][n] = att[n][m] / (eps + colsum[m]) as f16, 32x32 LDS transpose
__global__ __launch_bounds__(256) void att_transpose_norm(
    const float* __restrict__ att, const float* __restrict__ cs,
    _Float16* __restrict__ attT) {
  __shared__ float t[32][33];
  const int n0 = blockIdx.x * 32, m0 = blockIdx.y * 32;
  const int tx = threadIdx.x, ty = threadIdx.y;
  for (int r = ty; r < 32; r += 8)
    t[r][tx] = att[(size_t)(n0 + r) * NPTS + m0 + tx];
  __syncthreads();
  for (int r = ty; r < 32; r += 8)
    attT[(size_t)(m0 + r) * NPTS + n0 + tx] =
        (_Float16)(t[tx][r] / (1e-9f + cs[m0 + r]));
}

// yt[n][c] = h[n][c] - xr[c][n]  (f16), 32x32 LDS transpose of xr
__global__ __launch_bounds__(256) void yt_transpose(
    const float* __restrict__ xr, const float* __restrict__ hx,
    _Float16* __restrict__ yt, int C) {
  __shared__ float t[32][33];
  const int n0 = blockIdx.x * 32, c0 = blockIdx.y * 32;
  const int tx = threadIdx.x, ty = threadIdx.y;
  for (int r = ty; r < 32; r += 8)
    t[r][tx] = xr[(size_t)(c0 + r) * NPTS + n0 + tx];
  __syncthreads();
  for (int r = ty; r < 32; r += 8) {
    const size_t o = (size_t)(n0 + r) * C + c0 + tx;
    yt[o] = (_Float16)(hx[o] - t[tx][r]);
  }
}

__global__ void cvt_pad(const float* __restrict__ src,
                        _Float16* __restrict__ dst, int rows, int cols,
                        int dstRows) {
  const int i = blockIdx.x * 256 + threadIdx.x;
  if (i >= dstRows * cols) return;
  const int r = i / cols;
  dst[i] = (r < rows) ? (_Float16)src[i] : (_Float16)0.f;
}

__global__ void prep_xh0(const float* __restrict__ x,
                         _Float16* __restrict__ xh0) {
  const int i = blockIdx.x * 256 + threadIdx.x;
  if (i >= NBATCH * NPTS * 32) return;
  const int c = i & 31;
  const int n = i >> 5;
  xh0[i] = (c < 3) ? (_Float16)x[n * 3 + c] : (_Float16)0.f;
}

// max over points; ch 0..255 -> x4(256ch), 256..383 -> x3(128), 384..447 -> x2(64)
__global__ __launch_bounds__(256) void pool_max(const float* __restrict__ x4,
                                                const float* __restrict__ x3,
                                                const float* __restrict__ x2,
                                                float* __restrict__ f) {
  const int ch = blockIdx.x, b = blockIdx.y, tid = threadIdx.x;
  const float* src;
  int C;
  if (ch < 256) {
    src = x4 + (size_t)b * NPTS * 256 + ch;
    C = 256;
  } else if (ch < 384) {
    src = x3 + (size_t)b * NPTS * 128 + (ch - 256);
    C = 128;
  } else {
    src = x2 + (size_t)b * NPTS * 64 + (ch - 384);
    C = 64;
  }
  float m = -3.4e38f;
  for (int n = tid; n < NPTS; n += 256) m = fmaxf(m, src[(size_t)n * C]);
  __shared__ float red[256];
  red[tid] = m;
  __syncthreads();
  for (int s = 128; s > 0; s >>= 1) {
    if (tid < s) red[tid] = fmaxf(red[tid], red[tid + s]);
    __syncthreads();
  }
  if (tid == 0) f[b * 448 + ch] = red[0];
}

__global__ __launch_bounds__(256) void fc_head(
    const float* __restrict__ f, const float* w1, const float* b1,
    const float* g1, const float* s1, const float* w2, const float* b2,
    const float* g2, const float* s2, const float* w3, const float* b3,
    const float* g3, const float* s3, const float* w4, const float* b4,
    float* __restrict__ out) {
  __shared__ float a0[448], a1[256], a2[128], a3[16];
  const int tid = threadIdx.x;
  for (int b = 0; b < NBATCH; ++b) {
    for (int i = tid; i < 448; i += 256) a0[i] = f[b * 448 + i];
    __syncthreads();
    if (tid < 256) {
      float s = 0.f;
      for (int i = 0; i < 448; ++i) s += a0[i] * w1[tid * 448 + i];
      s = (s + b1[tid]) * g1[tid] + s1[tid];
      a1[tid] = fmaxf(s, 0.f);
    }
    __syncthreads();
    if (tid < 128) {
      float s = 0.f;
      for (int i = 0; i < 256; ++i) s += a1[i] * w2[tid * 256 + i];
      s = (s + b2[tid]) * g2[tid] + s2[tid];
      a2[tid] = fmaxf(s, 0.f);
    }
    __syncthreads();
    if (tid < 16) {
      float s = 0.f;
      for (int i = 0; i < 128; ++i) s += a2[i] * w3[tid * 128 + i];
      s = (s + b3[tid]) * g3[tid] + s3[tid];
      a3[tid] = fmaxf(s, 0.f);
    }
    __syncthreads();
    if (tid == 0) {
      float s = 0.f;
      for (int i = 0; i < 16; ++i) s += a3[i] * w4[i];
      out[b] = s + b4[0];
    }
    __syncthreads();
  }
}

// ---------------------------------------------------------------------------
extern "C" void kernel_launch(void* const* d_in, const int* in_sizes, int n_in,
                              void* d_out, int out_size, void* d_ws,
                              size_t ws_size, hipStream_t stream) {
  (void)in_sizes;
  (void)n_in;
  (void)out_size;
  (void)ws_size;
  const float* X = (const float*)d_in[0];
  auto F = [&](int i) { return (const float*)d_in[i]; };

  // ---- workspace carve-out ----
  char* base = (char*)d_ws;
  size_t off = 0;
  auto alloc = [&](size_t bytes) -> void* {
    void* p = base + off;
    off = (off + bytes + 255) & ~(size_t)255;
    return p;
  };
  const int Cout[4] = {64, 64, 128, 256};
  const int dpads[4] = {32, 32, 32, 64};
  const int drows[4] = {16, 16, 32, 64};
  const int abase[4] = {13, 20, 27, 34};  // att params start index

  _Float16* xh0 = (_Float16*)alloc((size_t)NBATCH * NPTS * 32 * 2);
  float* hF[4];
  _Float16* hH[4];
  float* xF[4];
  _Float16* xH[4];
  for (int l = 0; l < 4; ++l) {
    const size_t n = (size_t)NBATCH * NPTS * Cout[l];
    hF[l] = (float*)alloc(n * 4);
    hH[l] = (_Float16*)alloc(n * 2);
    xF[l] = (float*)alloc(n * 4);
    xH[l] = (_Float16*)alloc(n * 2);
  }
  float* gram = (float*)alloc((size_t)NPTS * NPTS * 4);
  float* dgb = (float*)alloc(NPTS * 4);
  float* csb = (float*)alloc(NPTS * 4);
  _Float16* attT = (_Float16*)alloc((size_t)NPTS * NPTS * 2);
  int* idxb = (int*)alloc((size_t)NBATCH * NPTS * KNN * 4);
  _Float16* qbuf = (_Float16*)alloc((size_t)NBATCH * NPTS * 64 * 2);
  _Float16* vbuf = (_Float16*)alloc((size_t)NBATCH * 256 * NPTS * 2);
  float* xrb = (float*)alloc((size_t)256 * NPTS * 4);
  _Float16* ytb = (_Float16*)alloc((size_t)NBATCH * NPTS * 256 * 2);
  float* fpool = (float*)alloc(NBATCH * 448 * 4);
  _Float16* wc[4] = {nullptr, nullptr, nullptr, nullptr};
  wc[1] = (_Float16*)alloc(64 * 128 * 2);
  wc[2] = (_Float16*)alloc(128 * 256 * 2);
  wc[3] = (_Float16*)alloc(256 * 512 * 2);
  _Float16 *qkh[4], *vwh[4], *twh[4];
  for (int l = 0; l < 4; ++l) {
    const int C = Cout[l];
    qkh[l] = (_Float16*)alloc((size_t)dpads[l] * C * 2);
    vwh[l] = (_Float16*)alloc((size_t)C * C * 2);
    twh[l] = (_Float16*)alloc((size_t)C * C * 2);
  }

  // ---- weight conversion f32 -> f16 (zero-padded where K/N not tile mult) --
  auto cvt = [&](const float* s, _Float16* d, int r, int c, int dr) {
    const int n = dr * c;
    cvt_pad<<<(n + 255) / 256, 256, 0, stream>>>(s, d, r, c, dr);
  };
  cvt(F(4), wc[1], 64, 128, 64);
  cvt(F(7), wc[2], 128, 256, 128);
  cvt(F(10), wc[3], 256, 512, 256);
  for (int l = 0; l < 4; ++l) {
    const int C = Cout[l], ab = abase[l];
    cvt(F(ab + 0), qkh[l], drows[l], C, dpads[l]);  // qk_w padded to dpad rows
    cvt(F(ab + 1), vwh[l], C, C, C);                // v_w
    cvt(F(ab + 3), twh[l], C, C, C);                // trans_w
  }
  prep_xh0<<<(NBATCH * NPTS * 32 + 255) / 256, 256, 0, stream>>>(X, xh0);

  // ---- 4 layers: knn -> edge conv -> offset self-attention ----
  for (int l = 0; l < 4; ++l) {
    const int C = Cout[l];
    const int Cin = (l == 0) ? 3 : Cout[l - 1];
    const int CinP = (l == 0) ? 32 : Cin;
    const _Float16* xin = (l == 0) ? xh0 : xH[l - 1];

    // kNN: Gram = xin @ xin^T (WMMA), diag, fused wave32 top-20. Per batch so
    // the 16MB NxN scratch stays a single L2-resident buffer.
    for (int b = 0; b < NBATCH; ++b) {
      const _Float16* xb = xin + (size_t)b * NPTS * CinP;
      gemm_big<1><<<dim3(NPTS / 64, NPTS / 128, 1), 128, 0, stream>>>(
          xb, xb, gram, nullptr, nullptr, nullptr, nullptr, nullptr, CinP,
          CinP, CinP, NPTS, 0, 0, 0);
      gram_diag<<<NPTS / 256, 256, 0, stream>>>(gram, dgb);
      knn_topk<<<NPTS / 8, 256, 0, stream>>>(gram, dgb,
                                             idxb + (size_t)b * NPTS * KNN);
    }

    // edge conv
    if (l == 0) {
      edge_conv1<<<(NBATCH * NPTS * 64 + 255) / 256, 256, 0, stream>>>(
          X, idxb, F(1), F(2), F(3), hF[0], hH[0]);
    } else {
      const float* eg = (l == 1) ? F(5) : (l == 2) ? F(8) : F(11);
      const float* eb = (l == 1) ? F(6) : (l == 2) ? F(9) : F(12);
      const size_t sh = (size_t)32 * (2 * Cin) * 2;
      edge_conv_wmma<<<dim3(NPTS / 64, C / 32, NBATCH), 128, sh, stream>>>(
          xin, idxb, wc[l], eg, eb, hF[l], hH[l], Cin, C);
    }

    // self-attention
    const int dp = dpads[l];
    const int ab = abase[l];
    // v = v_w @ h + v_b   -> (C, NPTS) f16, all batches
    gemm_big<2><<<dim3(C / 64, NPTS / 128, NBATCH), 128, 0, stream>>>(
        vwh[l], hH[l], nullptr, vbuf, F(ab + 2), nullptr, nullptr, nullptr, C,
        C, C, NPTS, 0, (long long)NPTS * C, (long long)C * NPTS);
    // q = h^T @ qk^T -> (NPTS, dp) f16 (pad cols auto-zero via padded weights)
    gemm16<0><<<dim3(NPTS / 64, dp / 32, NBATCH), 128, 0, stream>>>(
        hH[l], qkh[l], nullptr, qbuf, nullptr, nullptr, nullptr, nullptr, C, C,
        C, dp, (long long)NPTS * C, 0, (long long)NPTS * dp);
    for (int b = 0; b < NBATCH; ++b) {
      const _Float16* qb = qbuf + (size_t)b * NPTS * dp;
      // logits = q @ q^T (symmetric) -> gram f32
      gemm_big<1><<<dim3(NPTS / 64, NPTS / 128, 1), 128, 0, stream>>>(
          qb, qb, gram, nullptr, nullptr, nullptr, nullptr, nullptr, dp, dp,
          dp, NPTS, 0, 0, 0);
      softmax_row<<<NPTS, 256, 0, stream>>>(gram);
      col_sum<<<NPTS / 256, 256, 0, stream>>>(gram, csb);
      att_transpose_norm<<<dim3(NPTS / 32, NPTS / 32), dim3(32, 8), 0,
                           stream>>>(gram, csb, attT);
      // x_r = v @ att  -> (C, NPTS) f32
      gemm_big<1><<<dim3(C / 64, NPTS / 128, 1), 128, 0, stream>>>(
          vbuf + (size_t)b * C * NPTS, attT, xrb, nullptr, nullptr, nullptr,
          nullptr, nullptr, NPTS, NPTS, NPTS, NPTS, 0, 0, 0);
      yt_transpose<<<dim3(NPTS / 32, C / 32), dim3(32, 8), 0, stream>>>(
          xrb, hF[l] + (size_t)b * NPTS * C, ytb + (size_t)b * NPTS * C, C);
    }
    // x_out = h + relu((trans_w @ (h - x_r) + tb) * g + bb)
    gemm_big<3><<<dim3(C / 64, NPTS / 128, NBATCH), 128, 0, stream>>>(
        twh[l], ytb, xF[l], xH[l], F(ab + 4), F(ab + 5), F(ab + 6), hF[l], C,
        C, C, C, 0, (long long)NPTS * C, (long long)NPTS * C);
  }

  // ---- global max-pool concat + FC head ----
  pool_max<<<dim3(448, NBATCH), 256, 0, stream>>>(xF[3], xF[2], xF[1], fpool);
  fc_head<<<1, 256, 0, stream>>>(fpool, F(41), F(42), F(43), F(44), F(45),
                                 F(46), F(47), F(48), F(49), F(50), F(51),
                                 F(52), F(53), F(54), (float*)d_out);
}
